// StateSpaceLayer_60713657696872
// MI455X (gfx1250) — matrix operand order repaired and checked by reading, not compiled, not run on hardware
//
#include <hip/hip_runtime.h>
#include <hip/hip_bf16.h>

typedef __attribute__((ext_vector_type(16))) _Float16 v16h;
typedef __attribute__((ext_vector_type(8)))  float    v8f;
typedef __attribute__((ext_vector_type(4)))  float    v4f;

#define B_SZ 2
#define T_SZ 512
#define D_SZ 64

// ---------------------------------------------------------------------------
// Kernel 1: LayerNorm over (D,D) per (b,t), dt = softplus(xn . dt_w + dt_b),
// log_decay = dt * (-exp(log_A)).  Writes xnT as f16 in [b, r, c, t] layout
// (t contiguous) so kernel 3's WMMA B-operand loads are contiguous b128s,
// and log_decay in [b, r, t] layout (t contiguous) for the cumsum.
// Block = 256 threads: thread -> row r = tid>>2, 16 columns c0 = (tid&3)*16.
// ---------------------------------------------------------------------------
__global__ __launch_bounds__(256) void ln_dt_kernel(
    const float* __restrict__ x,      // [B,T,D,D]
    const float* __restrict__ log_A,  // [D]
    const float* __restrict__ dt_w,   // [D]
    const float* __restrict__ dt_b,   // [1]
    const float* __restrict__ ln_w,   // [D,D]
    const float* __restrict__ ln_b,   // [D,D]
    _Float16* __restrict__ xnT,       // [B,D,D,T]  (b,r,c,t)
    float* __restrict__ ld)           // [B,D,T]    (b,r,t)
{
    const int bt  = blockIdx.x;            // 0..B*T-1
    const int b   = bt >> 9;
    const int t   = bt & (T_SZ - 1);
    const int tid = threadIdx.x;
    const int r   = tid >> 2;
    const int c0  = (tid & 3) * 16;

    const float* xp = x + (size_t)bt * (D_SZ * D_SZ) + r * D_SZ + c0;

    float xv[16];
    float s = 0.f, ss = 0.f;
#pragma unroll
    for (int i = 0; i < 16; ++i) {
        float v = xp[i];
        xv[i] = v;
        s  += v;
        ss += v * v;
    }

    __shared__ float2 red[256];
    red[tid] = make_float2(s, ss);
    __syncthreads();
#pragma unroll
    for (int off = 128; off > 0; off >>= 1) {
        if (tid < off) {
            red[tid].x += red[tid + off].x;
            red[tid].y += red[tid + off].y;
        }
        __syncthreads();
    }
    const float inv_n = 1.0f / (float)(D_SZ * D_SZ);
    const float mu    = red[0].x * inv_n;
    const float var   = red[0].y * inv_n - mu * mu;
    const float rstd  = rsqrtf(var + 1e-5f);

    const float* wp = ln_w + r * D_SZ + c0;
    const float* bp = ln_b + r * D_SZ + c0;
    const float* dw = dt_w + c0;

    float xn[16];
    float dot = 0.f;
#pragma unroll
    for (int i = 0; i < 16; ++i) {
        float y = (xv[i] - mu) * rstd * wp[i] + bp[i];
        xn[i] = y;
        dot  += y * dw[i];
    }
    // reduce the row dot across the 4 aligned lanes covering this row
    dot += __shfl_xor(dot, 1, 32);
    dot += __shfl_xor(dot, 2, 32);

    if ((tid & 3) == 0) {
        float z  = dot + dt_b[0];
        float dt = (z > 20.f) ? z : log1pf(__expf(z));   // softplus
        float ac = -__expf(log_A[r]);
        ld[((size_t)(b * D_SZ + r)) * T_SZ + t] = dt * ac;
    }

    // scatter f16 xn into [b, r, c, t] layout
    _Float16* xo = xnT + (((size_t)(b * D_SZ + r)) * D_SZ + c0) * T_SZ + t;
#pragma unroll
    for (int i = 0; i < 16; ++i) {
        xo[(size_t)i * T_SZ] = (_Float16)xn[i];
    }
}

// ---------------------------------------------------------------------------
// Kernel 2: inclusive cumsum over t for each of the B*D=128 series.
// One thread per (b,r); contiguous 512-element walk. Negligible cost.
// ---------------------------------------------------------------------------
__global__ __launch_bounds__(128) void cumsum_kernel(
    const float* __restrict__ ld,   // [B,D,T]
    float* __restrict__ cs)         // [B,D,T]
{
    const int idx = threadIdx.x;    // 0..127
    const float* in = ld + (size_t)idx * T_SZ;
    float* o        = cs + (size_t)idx * T_SZ;
    float acc = 0.f;
    for (int t = 0; t < T_SZ; ++t) {
        acc += in[t];
        o[t] = acc;
    }
}

// ---------------------------------------------------------------------------
// Kernel 3: out[b,t,r,c] = sum_{j<=t} exp(cs[t]-cs[j]) * xn[b,j,r,c]
// via v_wmma_f32_16x16x32_f16.  Grid = (T/64, D, B), 128 threads = 4 waves.
// Wave w owns t-rows [tt*64 + w*16, +16) and 4 c-subtiles (full 64 columns).
// A tile (16x32 f16) is built branchlessly: the causal mask selects the exp
// ARGUMENT (-1e30 -> exp==0) so no EXEC-mask branching occurs and the
// v_exp_f32 TRANS ops co-execute with WMMA.  cs reads are 4x ds_load_b128.
// B tiles (32x16 f16) are two aligned 16B global loads per c-subtile from
// the [b,r,c,t] xnT layout.  Loop bound is wave-uniform -> EXEC all 1s.
// ---------------------------------------------------------------------------
__global__ __launch_bounds__(128) void ssm_attn_wmma(
    const _Float16* __restrict__ xnT,  // [B,D,D,T]
    const float* __restrict__ cs,      // [B,D,T]
    float* __restrict__ out)           // [B,T,D,D]
{
    const int tt = blockIdx.x;   // t-tile of 64 rows: 0..7
    const int r  = blockIdx.y;   // 0..63
    const int b  = blockIdx.z;   // 0..1

    __shared__ float cs_s[T_SZ];
    const float* csr = cs + ((size_t)(b * D_SZ + r)) * T_SZ;
    for (int i = threadIdx.x; i < T_SZ; i += 128) cs_s[i] = csr[i];
    __syncthreads();

    const int wave  = threadIdx.x >> 5;
    const int lane  = threadIdx.x & 31;
    const int hi    = lane >> 4;     // lane half (selects K sub-range)
    const int l16   = lane & 15;
    const int tbase = tt * 64 + wave * 16;
    const int trow  = tbase + l16;   // this lane's A-matrix row (t index)
    const float cst = cs_s[trow];

    const _Float16* xb = xnT + ((size_t)(b * D_SZ + r)) * D_SZ * T_SZ;

    v8f acc0 = {}, acc1 = {}, acc2 = {}, acc3 = {};

    const int jend = tbase + 16;               // wave-uniform causal bound
    for (int j0 = 0; j0 < jend; j0 += 32) {
        // Lane's 16 K-values are two runs of 8 consecutive j's:
        //   e=0..7  -> j = kA + e,  kA = j0 + hi*8
        //   e=8..15 -> j = kB + e-8, kB = j0 + 16 + hi*8
        const int kA = j0 + hi * 8;
        const int kB = j0 + 16 + hi * 8;
        v4f cj0 = *(const v4f*)&cs_s[kA];
        v4f cj1 = *(const v4f*)&cs_s[kA + 4];
        v4f cj2 = *(const v4f*)&cs_s[kB];
        v4f cj3 = *(const v4f*)&cs_s[kB + 4];
        float cj[16];
#pragma unroll
        for (int q = 0; q < 4; ++q) {
            cj[q]      = cj0[q];
            cj[q + 4]  = cj1[q];
            cj[q + 8]  = cj2[q];
            cj[q + 12] = cj3[q];
        }

        v16h a;
#pragma unroll
        for (int e = 0; e < 16; ++e) {
            const int j = (e < 8) ? (kA + e) : (kB + e - 8);
            // branchless causal mask: masked lanes get -1e30 -> exp -> 0
            const float arg = (j <= trow) ? (cst - cj[e]) : -1e30f;
            a[e] = (_Float16)__expf(arg);
        }

        // ---- Load B (32x16 f16): lane half hi holds K=8hi..8hi+7 (e<8)
        //      and K=16+8hi..16+8hi+7 (e>=8) of column l16, matching A.
        //      Contiguous along t(j) in the [b,r,c,t] layout -> 2x b128. ----
        const _Float16* bp = xb + (size_t)l16 * T_SZ + (j0 + hi * 8);
        typedef __attribute__((ext_vector_type(8))) _Float16 v8h_;
        auto ldb = [&](const _Float16* p) -> v16h {
            v8h_ lo = *(const v8h_*)(p);
            v8h_ up = *(const v8h_*)(p + 16);
            return __builtin_shufflevector(lo, up, 0,1,2,3,4,5,6,7,8,9,10,11,12,13,14,15);
        };
        v16h b0 = ldb(bp);
        v16h b1 = ldb(bp + 16 * T_SZ);
        v16h b2 = ldb(bp + 32 * T_SZ);
        v16h b3 = ldb(bp + 48 * T_SZ);

        acc0 = __builtin_amdgcn_wmma_f32_16x16x32_f16(false, a, false, b0,
                                                      (short)0, acc0, false, false);
        acc1 = __builtin_amdgcn_wmma_f32_16x16x32_f16(false, a, false, b1,
                                                      (short)0, acc1, false, false);
        acc2 = __builtin_amdgcn_wmma_f32_16x16x32_f16(false, a, false, b2,
                                                      (short)0, acc2, false, false);
        acc3 = __builtin_amdgcn_wmma_f32_16x16x32_f16(false, a, false, b3,
                                                      (short)0, acc3, false, false);
    }

    // ---- Store C: VGPR v holds row (v + 8*hi), col = l16 (+16*subtile). ----
    float* ob = out + (((size_t)(b * T_SZ + tbase)) * D_SZ + r) * D_SZ;
#pragma unroll
    for (int v = 0; v < 8; ++v) {
        const int row = hi * 8 + v;
        float* orow = ob + (size_t)row * (D_SZ * D_SZ) + l16;
        orow[0]  = acc0[v];
        orow[16] = acc1[v];
        orow[32] = acc2[v];
        orow[48] = acc3[v];
    }
}

// ---------------------------------------------------------------------------
extern "C" void kernel_launch(void* const* d_in, const int* in_sizes, int n_in,
                              void* d_out, int out_size, void* d_ws, size_t ws_size,
                              hipStream_t stream) {
    const float* x     = (const float*)d_in[0];   // [2,512,64,64]
    const float* log_A = (const float*)d_in[1];   // [64]
    const float* dt_w  = (const float*)d_in[2];   // [64]
    const float* dt_b  = (const float*)d_in[3];   // [1]
    const float* ln_w  = (const float*)d_in[4];   // [64,64]
    const float* ln_b  = (const float*)d_in[5];   // [64,64]
    float* out = (float*)d_out;                   // [2,512,64,64]

    char* ws = (char*)d_ws;
    const size_t xnT_bytes = (size_t)B_SZ * D_SZ * D_SZ * T_SZ * sizeof(_Float16); // 8 MB
    const size_t ld_bytes  = (size_t)B_SZ * D_SZ * T_SZ * sizeof(float);           // 256 KB
    _Float16* xnT = (_Float16*)ws;
    float*    ld  = (float*)(ws + xnT_bytes);
    float*    cs  = (float*)(ws + xnT_bytes + ld_bytes);

    ln_dt_kernel<<<B_SZ * T_SZ, 256, 0, stream>>>(x, log_A, dt_w, dt_b, ln_w, ln_b,
                                                  xnT, ld);
    cumsum_kernel<<<1, 128, 0, stream>>>(ld, cs);

    dim3 g3(T_SZ / 64, D_SZ, B_SZ);
    ssm_attn_wmma<<<g3, 128, 0, stream>>>(xnT, cs, out);
}
